// LongformerAttention_7791070675575
// MI455X (gfx1250) — matrix operand order repaired and checked by reading, not compiled, not run on hardware
//
#include <hip/hip_runtime.h>

#define T_LEN   4096
#define D_MODEL 512
#define NUM_H   8
#define HEAD_D  64
#define WIN     256
#define NBIAS   (2 * WIN - 1)   // 511
#define LDST    72              // padded LDS row stride (halves) to spread banks

typedef __attribute__((ext_vector_type(16))) _Float16 v16h;
typedef __attribute__((ext_vector_type(8)))  _Float16 v8h;
typedef __attribute__((ext_vector_type(8)))  float    v8f;

// Build a 16-half fragment from two 8-half (16B) chunks.
static __device__ __forceinline__ v16h frag2(const _Float16* p0, const _Float16* p1) {
    v8h a = *(const v8h*)p0;
    v8h b = *(const v8h*)p1;
    return __builtin_shufflevector(a, b, 0,1,2,3,4,5,6,7,8,9,10,11,12,13,14,15);
}

// ---------------------------------------------------------------------------
// f32 -> f16 conversion (grid-stride)
// ---------------------------------------------------------------------------
__global__ void cvt_f16_kernel(const float* __restrict__ in,
                               _Float16* __restrict__ out, int n) {
    int i = blockIdx.x * blockDim.x + threadIdx.x;
    int stride = gridDim.x * blockDim.x;
    for (; i < n; i += stride) out[i] = (_Float16)in[i];
}

// ---------------------------------------------------------------------------
// Projection GEMM: Y[t][o] = sum_k X[t][k] * W[o][k]    (y = x @ W^T)
// X: (T x D) f16 row-major, W: (D x D) f16 row-major (out-major).
// TRANSPOSED=false -> Y[t*D + o] ; TRANSPOSED=true -> Y[o*T + t] (for V)
// Block = 128 threads (4 waves). Block tile = 64(t) x 64(o). Wave = 16 t-rows.
// ---------------------------------------------------------------------------
template <bool TRANSPOSED>
__global__ void __launch_bounds__(128)
proj_kernel(const _Float16* __restrict__ X,
            const _Float16* __restrict__ W,
            _Float16* __restrict__ Y) {
    const int lane  = threadIdx.x & 31;
    const int wave  = threadIdx.x >> 5;           // 0..3
    const int m     = lane & 15;                  // A row / B col / C col
    const int kca   = (lane >> 4) * 8;            // A-frag K chunk base
    const int kcb   = (lane >> 4) * 16;           // unused; B-frag uses kca as well
    const int rhalf = (lane >> 4) * 8;            // C row offset
    const int tbase = blockIdx.x * 64 + wave * 16;
    const int obase = blockIdx.y * 64;

    v8f acc[4] = {};

    for (int k0 = 0; k0 < D_MODEL; k0 += 32) {
        const _Float16* xr = X + (size_t)(tbase + m) * D_MODEL + k0;
        v16h a = frag2(xr + kca, xr + kca + 16);
#pragma unroll
        for (int nt = 0; nt < 4; ++nt) {
            const _Float16* wr = W + (size_t)(obase + nt * 16 + m) * D_MODEL + k0;
            v16h b = frag2(wr + kca, wr + kca + 16);
            acc[nt] = __builtin_amdgcn_wmma_f32_16x16x32_f16(
                false, a, false, b, (short)0, acc[nt], false, false);
        }
    }

#pragma unroll
    for (int nt = 0; nt < 4; ++nt) {
        if (TRANSPOSED) {
            // fixed output feature o, 8 consecutive t -> one 16B store
            int o = obase + nt * 16 + m;
            v8h pk;
#pragma unroll
            for (int r = 0; r < 8; ++r) pk[r] = (_Float16)acc[nt][r];
            *(v8h*)(Y + (size_t)o * T_LEN + tbase + rhalf) = pk;
        } else {
#pragma unroll
            for (int r = 0; r < 8; ++r) {
                int t = tbase + rhalf + r;
                Y[(size_t)t * D_MODEL + obase + nt * 16 + m] = (_Float16)acc[nt][r];
            }
        }
    }
}

// ---------------------------------------------------------------------------
// Flash attention with rel-pos bias, register-pipelined K/V staging.
// Grid: (T/64, H). Block = 128 threads (4 waves); wave = 16 query rows.
// Q,K: (T x D) f16 (head slice at h*64). Vt: (D x T) f16 transposed.
// ---------------------------------------------------------------------------
__global__ void __launch_bounds__(128)
attn_kernel(const _Float16* __restrict__ Q,
            const _Float16* __restrict__ K,
            const _Float16* __restrict__ Vt,
            const float* __restrict__ relBias,
            float* __restrict__ Out) {
    __shared__ __align__(16) _Float16 sK[64 * LDST];
    __shared__ __align__(16) _Float16 sV[64 * LDST];
    __shared__ __align__(16) _Float16 sP[4][16 * LDST];
    __shared__ float sBias[NBIAS];

    const int tid   = threadIdx.x;
    const int lane  = tid & 31;
    const int wave  = tid >> 5;
    const int m     = lane & 15;
    const int kca   = (lane >> 4) * 8;
    const int kcb   = (lane >> 4) * 16;
    const int rhalf = (lane >> 4) * 8;
    const int h     = blockIdx.y;
    const int qbase = blockIdx.x * 64 + wave * 16;

    // stage this head's bias table
    for (int i = tid; i < NBIAS; i += 128) sBias[i] = relBias[h * NBIAS + i];

    // Q fragments held in registers for the whole kernel (hd = 64 -> 2 frags)
    const _Float16* qr = Q + (size_t)(qbase + m) * D_MODEL + h * HEAD_D;
    v16h qf0 = frag2(qr + kca,      qr + kca + 16);
    v16h qf1 = frag2(qr + 32 + kca, qr + 48 + kca);

    const float scale = 0.125f;  // 1/sqrt(64)

    v8f   oacc[4] = {};
    float mrow[8], lrow[8];
#pragma unroll
    for (int r = 0; r < 8; ++r) { mrow[r] = -1e30f; lrow[r] = 0.f; }

    // each thread stages 4 x 16B of K and of Vt per tile, via registers
    const int srow = tid >> 3;            // 0..15 (rows tid covers: srow + i*16)
    const int scol = (tid & 7) * 8;       // half offset within row

    v8h kreg[4], vreg[4];
#pragma unroll
    for (int i = 0; i < 4; ++i) {
        int row = srow + i * 16;
        kreg[i] = *(const v8h*)(K  + (size_t)row * D_MODEL + h * HEAD_D + scol);
        vreg[i] = *(const v8h*)(Vt + (size_t)(h * HEAD_D + row) * T_LEN + scol);
    }

    for (int kb = 0; kb < T_LEN; kb += 64) {
        __syncthreads();    // all waves done reading previous tile
#pragma unroll
        for (int i = 0; i < 4; ++i) {
            int row = srow + i * 16;
            *(v8h*)&sK[row * LDST + scol] = kreg[i];
            *(v8h*)&sV[row * LDST + scol] = vreg[i];
        }
        __syncthreads();    // tile visible to all waves

        // prefetch next tile into registers; retires under the compute below
        if (kb + 64 < T_LEN) {
#pragma unroll
            for (int i = 0; i < 4; ++i) {
                int row = srow + i * 16;
                kreg[i] = *(const v8h*)(K + (size_t)(kb + 64 + row) * D_MODEL +
                                        h * HEAD_D + scol);
                vreg[i] = *(const v8h*)(Vt + (size_t)(h * HEAD_D + row) * T_LEN +
                                        kb + 64 + scol);
            }
        }

        // ---- S = scale * Q K^T + bias  (4 j-tiles of 16) ----
        float sbuf[4][8];
        float mcur[8];
#pragma unroll
        for (int r = 0; r < 8; ++r) mcur[r] = -1e30f;

#pragma unroll
        for (int jt = 0; jt < 4; ++jt) {
            const _Float16* krow = &sK[(jt * 16 + m) * LDST];
            v16h kf0 = frag2(krow + kca,      krow + kca + 16);
            v16h kf1 = frag2(krow + 32 + kca, krow + 48 + kca);
            v8f s = {};
            s = __builtin_amdgcn_wmma_f32_16x16x32_f16(false, qf0, false, kf0,
                                                       (short)0, s, false, false);
            s = __builtin_amdgcn_wmma_f32_16x16x32_f16(false, qf1, false, kf1,
                                                       (short)0, s, false, false);
            int j = kb + jt * 16 + m;
#pragma unroll
            for (int r = 0; r < 8; ++r) {
                int i   = qbase + rhalf + r;
                int rel = i - j;
                rel = rel < -(WIN - 1) ? -(WIN - 1)
                                       : (rel > (WIN - 1) ? (WIN - 1) : rel);
                float sv = s[r] * scale + sBias[rel + (WIN - 1)];
                sbuf[jt][r] = sv;
                mcur[r] = fmaxf(mcur[r], sv);
            }
        }

        // row max across the 16 lanes holding a row
#pragma unroll
        for (int r = 0; r < 8; ++r) {
#pragma unroll
            for (int off = 1; off < 16; off <<= 1)
                mcur[r] = fmaxf(mcur[r], __shfl_xor(mcur[r], off, 32));
        }

        float alpha[8];
#pragma unroll
        for (int r = 0; r < 8; ++r) {
            float mn = fmaxf(mrow[r], mcur[r]);
            alpha[r] = __expf(mrow[r] - mn);
            mrow[r]  = mn;
        }

        // p = exp(s - m); write P tile (16 x 64 f16) to this wave's LDS region
        float rsum[8];
#pragma unroll
        for (int r = 0; r < 8; ++r) rsum[r] = 0.f;
#pragma unroll
        for (int jt = 0; jt < 4; ++jt) {
#pragma unroll
            for (int r = 0; r < 8; ++r) {
                float p = __expf(sbuf[jt][r] - mrow[r]);
                rsum[r] += p;
                sP[wave][(rhalf + r) * LDST + jt * 16 + m] = (_Float16)p;
            }
        }
#pragma unroll
        for (int r = 0; r < 8; ++r) {
#pragma unroll
            for (int off = 1; off < 16; off <<= 1)
                rsum[r] += __shfl_xor(rsum[r], off, 32);
            lrow[r] = lrow[r] * alpha[r] + rsum[r];
        }

        // rescale O accumulators
#pragma unroll
        for (int dt = 0; dt < 4; ++dt)
#pragma unroll
            for (int r = 0; r < 8; ++r) oacc[dt][r] *= alpha[r];

        // ---- O += P V  (A = P from LDS, B = Vt rows from LDS) ----
        const _Float16* prow = &sP[wave][m * LDST];
        v16h pf0 = frag2(prow + kca,      prow + kca + 16);
        v16h pf1 = frag2(prow + 32 + kca, prow + 48 + kca);
#pragma unroll
        for (int dt = 0; dt < 4; ++dt) {
            const _Float16* vrow = &sV[(dt * 16 + m) * LDST];
            v16h vf0 = frag2(vrow + kca,      vrow + kca + 16);
            v16h vf1 = frag2(vrow + 32 + kca, vrow + 48 + kca);
            oacc[dt] = __builtin_amdgcn_wmma_f32_16x16x32_f16(
                false, pf0, false, vf0, (short)0, oacc[dt], false, false);
            oacc[dt] = __builtin_amdgcn_wmma_f32_16x16x32_f16(
                false, pf1, false, vf1, (short)0, oacc[dt], false, false);
        }
    }

    // epilogue: normalize and store f32 output (B,T,D) with head slice h*64
#pragma unroll
    for (int dt = 0; dt < 4; ++dt) {
#pragma unroll
        for (int r = 0; r < 8; ++r) {
            int t = qbase + rhalf + r;
            Out[(size_t)t * D_MODEL + h * HEAD_D + dt * 16 + m] =
                oacc[dt][r] / lrow[r];
        }
    }
}

// ---------------------------------------------------------------------------
extern "C" void kernel_launch(void* const* d_in, const int* in_sizes, int n_in,
                              void* d_out, int out_size, void* d_ws, size_t ws_size,
                              hipStream_t stream) {
    const float* x  = (const float*)d_in[0];
    const float* Wq = (const float*)d_in[1];
    const float* Wk = (const float*)d_in[2];
    const float* Wv = (const float*)d_in[3];
    const float* rb = (const float*)d_in[4];
    float* out = (float*)d_out;

    _Float16* ws  = (_Float16*)d_ws;
    const size_t TD = (size_t)T_LEN * D_MODEL;    // 2M halves
    const size_t DD = (size_t)D_MODEL * D_MODEL;  // 256K halves
    _Float16* xh  = ws;
    _Float16* qh  = xh  + TD;
    _Float16* kh  = qh  + TD;
    _Float16* vt  = kh  + TD;
    _Float16* wqh = vt  + TD;
    _Float16* wkh = wqh + DD;
    _Float16* wvh = wkh + DD;

    cvt_f16_kernel<<<1024, 256, 0, stream>>>(x,  xh,  (int)TD);
    cvt_f16_kernel<<<512,  256, 0, stream>>>(Wq, wqh, (int)DD);
    cvt_f16_kernel<<<512,  256, 0, stream>>>(Wk, wkh, (int)DD);
    cvt_f16_kernel<<<512,  256, 0, stream>>>(Wv, wvh, (int)DD);

    dim3 pgrid(T_LEN / 64, D_MODEL / 64);
    proj_kernel<false><<<pgrid, 128, 0, stream>>>(xh, wqh, qh);
    proj_kernel<false><<<pgrid, 128, 0, stream>>>(xh, wkh, kh);
    proj_kernel<true ><<<pgrid, 128, 0, stream>>>(xh, wvh, vt);

    dim3 agrid(T_LEN / 64, NUM_H);
    attn_kernel<<<agrid, 128, 0, stream>>>(qh, kh, vt, rb, out);
}
